// rgcn_Layers_15599321219706
// MI455X (gfx1250) — compile-verified, matrix operand order checked by reading, not run on hardware
//
#include <hip/hip_runtime.h>

typedef __attribute__((ext_vector_type(16))) __bf16 v16bf;
typedef __attribute__((ext_vector_type(8)))  __bf16 v8bf;
typedef __attribute__((ext_vector_type(4)))  __bf16 v4bf;
typedef __attribute__((ext_vector_type(8)))  float  v8f;

#define NN   50000
#define EE   1600000
#define RR   8
#define IN_D 514
#define KP1  544      // 514 padded to 17*32
#define HD   128
#define LD   32
#define KT1  17
#define NT1  8
#define KT2  4
#define NT2  2
#define MT   3125     // 50000/16 exact

// ---------------- elementwise / prep kernels ----------------

__global__ __launch_bounds__(256) void k_cvt_emb(const float* __restrict__ emb,
                                                 __bf16* __restrict__ out) {
  size_t i = (size_t)blockIdx.x * 256 + threadIdx.x;
  if (i >= (size_t)NN * KP1) return;
  int k = (int)(i % KP1);
  size_t row = i / KP1;
  float v = (k < IN_D) ? emb[row * IN_D + k] : 0.0f;
  out[i] = (__bf16)v;
}

// Pack weights into per-lane WMMA B-operand order:
// tile t = (z*NT+nt)*KT+kt ; within tile: lane*16 + i
// k-offset per (i, half) follows the 16-bit A/B VGPR layout table.
__global__ __launch_bounds__(512) void k_packW1(const float* __restrict__ W1,
                                                const float* __restrict__ root1,
                                                __bf16* __restrict__ Bp) {
  int t = blockIdx.x;
  int kt = t % KT1; int zn = t / KT1; int nt = zn % NT1; int z = zn / NT1;
  int tid = threadIdx.x;
  int lane = tid >> 4, i = tid & 15;
  int c = lane & 15, half = lane >> 4;
  int k = kt * 32 + i + (i < 8 ? (half ? 8 : 0) : (half ? 16 : 8));
  int n = nt * 16 + c;
  float v = 0.0f;
  if (k < IN_D)
    v = (z < RR) ? W1[((size_t)z * IN_D + k) * HD + n] : root1[(size_t)k * HD + n];
  Bp[(size_t)t * 512 + tid] = (__bf16)v;
}

__global__ __launch_bounds__(512) void k_packW2(const float* __restrict__ W2,
                                                const float* __restrict__ root2,
                                                __bf16* __restrict__ Bp) {
  int t = blockIdx.x;
  int kt = t % KT2; int zn = t / KT2; int nt = zn % NT2; int z = zn / NT2;
  int tid = threadIdx.x;
  int lane = tid >> 4, i = tid & 15;
  int c = lane & 15, half = lane >> 4;
  int k = kt * 32 + i + (i < 8 ? (half ? 8 : 0) : (half ? 16 : 8));
  int n = nt * 16 + c;
  float v = (z < RR) ? W2[((size_t)z * HD + k) * LD + n] : root2[(size_t)k * LD + n];
  Bp[(size_t)t * 512 + tid] = (__bf16)v;
}

__global__ __launch_bounds__(256) void k_zero_cnt(int* __restrict__ cnt) {
  int i = blockIdx.x * 256 + threadIdx.x;
  if (i < NN * RR) cnt[i] = 0;
}

__global__ __launch_bounds__(256) void k_count(const int* __restrict__ dst,
                                               const int* __restrict__ et,
                                               int* __restrict__ cnt) {
  int e = blockIdx.x * 256 + threadIdx.x;
  if (e < EE) atomicAdd(&cnt[dst[e] * RR + et[e]], 1);
}

__global__ __launch_bounds__(256) void k_inv(const int* __restrict__ cnt,
                                             float* __restrict__ invc) {
  int i = blockIdx.x * 256 + threadIdx.x;
  if (i < NN * RR) invc[i] = 1.0f / fmaxf((float)cnt[i], 1.0f);
}

// ---------------- WMMA GEMMs ----------------
// One wave per 16-row M-tile, full N per wave. grid.z = matrix index (0..7 rel, 8 root).
// gemm1: B staged in LDS (shared by all 8 waves of the block), A loads pipelined.

__global__ __launch_bounds__(256) void k_gemm1(const __bf16* __restrict__ A,   // [NN][KP1]
                                               const __bf16* __restrict__ Bp,  // packed
                                               const float* __restrict__ bias1,
                                               __bf16* __restrict__ h1,        // [RR][NN][HD]
                                               float* __restrict__ x1acc) {    // [NN][HD]
  __shared__ __bf16 Bs[NT1 * KT1 * 512];   // 139,264 bytes (fits in 320KB WGP LDS)

  // Cooperative stage of this z's packed B into LDS (all waves share z).
  {
    const __bf16* gsrc = Bp + (size_t)blockIdx.z * NT1 * KT1 * 512;
    for (int idx = threadIdx.x; idx < NT1 * KT1 * 512 / 16; idx += 256) {
      *(v16bf*)(Bs + (size_t)idx * 16) = *(const v16bf*)(gsrc + (size_t)idx * 16);
    }
  }
  __syncthreads();

  const int wave = threadIdx.x >> 5, lane = threadIdx.x & 31;
  const int mt = blockIdx.x * 8 + wave;
  if (mt >= MT) return;
  const int z = blockIdx.z;
  const int c = lane & 15, half = lane >> 4;
  v8f acc[NT1] = {};
  const __bf16* Arow = A + (size_t)(mt * 16 + c) * KP1 + half * 8;
  const __bf16* Bl = Bs + (size_t)lane * 16;

  v8bf a0 = *(const v8bf*)(Arow);
  v8bf a1 = *(const v8bf*)(Arow + 16);
  for (int kt = 0; kt < KT1; ++kt) {
    v8bf na0, na1;
    if (kt + 1 < KT1) {              // prefetch next A pair while computing
      na0 = *(const v8bf*)(Arow + (kt + 1) * 32);
      na1 = *(const v8bf*)(Arow + (kt + 1) * 32 + 16);
    }
    v16bf a = __builtin_shufflevector(a0, a1, 0,1,2,3,4,5,6,7,8,9,10,11,12,13,14,15);
#pragma unroll
    for (int nt = 0; nt < NT1; ++nt) {
      v16bf b = *(const v16bf*)(Bl + (size_t)(nt * KT1 + kt) * 512);
      acc[nt] = __builtin_amdgcn_wmma_f32_16x16x32_bf16(
          false, a, false, b, (short)0, acc[nt], false, false);
    }
    a0 = na0; a1 = na1;
  }

  if (z < RR) {
    __bf16* out = h1 + (size_t)z * NN * HD;
#pragma unroll
    for (int nt = 0; nt < NT1; ++nt) {
      int col = nt * 16 + c;
#pragma unroll
      for (int j = 0; j < 8; ++j) {
        int row = mt * 16 + half * 8 + j;
        out[(size_t)row * HD + col] = (__bf16)acc[nt][j];
      }
    }
  } else {
#pragma unroll
    for (int nt = 0; nt < NT1; ++nt) {
      int col = nt * 16 + c;
      float bv = bias1[col];
#pragma unroll
      for (int j = 0; j < 8; ++j) {
        int row = mt * 16 + half * 8 + j;
        x1acc[(size_t)row * HD + col] = acc[nt][j] + bv;
      }
    }
  }
}

// gemm2: entire B for this z (8 tiles = 8KB) preloaded into registers.
__global__ __launch_bounds__(256) void k_gemm2(const __bf16* __restrict__ A,   // [NN][HD]
                                               const __bf16* __restrict__ Bp,
                                               const float* __restrict__ bias2,
                                               __bf16* __restrict__ g2,        // [RR][NN][LD]
                                               float* __restrict__ oacc) {     // [NN][LD]
  const int wave = threadIdx.x >> 5, lane = threadIdx.x & 31;
  const int mt = blockIdx.x * 8 + wave;
  if (mt >= MT) return;
  const int z = blockIdx.z;
  const int c = lane & 15, half = lane >> 4;

  const __bf16* Bz = Bp + (size_t)z * NT2 * KT2 * 512 + (size_t)lane * 16;
  v16bf breg[NT2 * KT2];
#pragma unroll
  for (int t = 0; t < NT2 * KT2; ++t)
    breg[t] = *(const v16bf*)(Bz + (size_t)t * 512);   // t = nt*KT2 + kt

  v8f acc[NT2] = {};
  const __bf16* Arow = A + (size_t)(mt * 16 + c) * HD + half * 8;
#pragma unroll
  for (int kt = 0; kt < KT2; ++kt) {
    v8bf a0 = *(const v8bf*)(Arow + kt * 32);
    v8bf a1 = *(const v8bf*)(Arow + kt * 32 + 16);
    v16bf a = __builtin_shufflevector(a0, a1, 0,1,2,3,4,5,6,7,8,9,10,11,12,13,14,15);
#pragma unroll
    for (int nt = 0; nt < NT2; ++nt) {
      acc[nt] = __builtin_amdgcn_wmma_f32_16x16x32_bf16(
          false, a, false, breg[nt * KT2 + kt], (short)0, acc[nt], false, false);
    }
  }
  if (z < RR) {
    __bf16* out = g2 + (size_t)z * NN * LD;
#pragma unroll
    for (int nt = 0; nt < NT2; ++nt) {
      int col = nt * 16 + c;
#pragma unroll
      for (int j = 0; j < 8; ++j) {
        int row = mt * 16 + half * 8 + j;
        out[(size_t)row * LD + col] = (__bf16)acc[nt][j];
      }
    }
  } else {
#pragma unroll
    for (int nt = 0; nt < NT2; ++nt) {
      int col = nt * 16 + c;
      float bv = bias2[col];
#pragma unroll
      for (int j = 0; j < 8; ++j) {
        int row = mt * 16 + half * 8 + j;
        oacc[(size_t)row * LD + col] = acc[nt][j] + bv;
      }
    }
  }
}

// ---------------- edge scatter (mean aggregation, pre-scaled) ----------------

__global__ __launch_bounds__(256) void k_scatter1(const int* __restrict__ src,
                                                  const int* __restrict__ dst,
                                                  const int* __restrict__ et,
                                                  const __bf16* __restrict__ h1,
                                                  const float* __restrict__ invc,
                                                  float* __restrict__ x1acc) {
  int e = (int)((blockIdx.x * 256u + threadIdx.x) >> 5);
  if (e >= EE) return;
  int lane = threadIdx.x & 31;
  int s = src[e], d = dst[e], r = et[e];
  float w = invc[d * RR + r];
  v4bf hv = *(const v4bf*)(h1 + ((size_t)r * NN + s) * HD + lane * 4);
  float* op = x1acc + (size_t)d * HD + lane * 4;
#pragma unroll
  for (int i = 0; i < 4; ++i) unsafeAtomicAdd(op + i, (float)hv[i] * w);
}

__global__ __launch_bounds__(256) void k_scatter2(const int* __restrict__ src,
                                                  const int* __restrict__ dst,
                                                  const int* __restrict__ et,
                                                  const __bf16* __restrict__ g2,
                                                  const float* __restrict__ invc,
                                                  float* __restrict__ oacc) {
  int e = (int)((blockIdx.x * 256u + threadIdx.x) >> 5);
  if (e >= EE) return;
  int lane = threadIdx.x & 31;
  int s = src[e], d = dst[e], r = et[e];
  float w = invc[d * RR + r];
  float v = (float)g2[((size_t)r * NN + s) * LD + lane];
  unsafeAtomicAdd(oacc + (size_t)d * LD + lane, v * w);
}

__global__ __launch_bounds__(256) void k_relu(const float* __restrict__ x,
                                              __bf16* __restrict__ xb) {
  int i = blockIdx.x * 256 + threadIdx.x;
  if (i < NN * HD) xb[i] = (__bf16)fmaxf(x[i], 0.0f);
}

__global__ __launch_bounds__(256) void k_sig(const float* __restrict__ acc,
                                             float* __restrict__ out) {
  int i = blockIdx.x * 256 + threadIdx.x;
  if (i < NN * LD) out[i] = 1.0f / (1.0f + __expf(-acc[i]));
}

// ---------------- launcher ----------------

extern "C" void kernel_launch(void* const* d_in, const int* in_sizes, int n_in,
                              void* d_out, int out_size, void* d_ws, size_t ws_size,
                              hipStream_t stream) {
  const int*   edge_index = (const int*)d_in[0];
  const int*   edge_type  = (const int*)d_in[1];
  const float* emb   = (const float*)d_in[2];
  const float* W1    = (const float*)d_in[3];
  const float* root1 = (const float*)d_in[4];
  const float* bias1 = (const float*)d_in[5];
  const float* W2    = (const float*)d_in[6];
  const float* root2 = (const float*)d_in[7];
  const float* bias2 = (const float*)d_in[8];
  const int* src  = edge_index;
  const int* dstv = edge_index + EE;

  char* p = (char*)d_ws;
  auto carve = [&](size_t bytes) -> char* {
    char* q = p; p += (bytes + 255) & ~(size_t)255; return q;
  };
  __bf16* embb  = (__bf16*)carve((size_t)NN * KP1 * 2);
  __bf16* W1p   = (__bf16*)carve((size_t)9 * NT1 * KT1 * 512 * 2);
  __bf16* W2p   = (__bf16*)carve((size_t)9 * NT2 * KT2 * 512 * 2);
  __bf16* h1    = (__bf16*)carve((size_t)RR * NN * HD * 2);
  float*  x1acc = (float*) carve((size_t)NN * HD * 4);
  int*    cnt   = (int*)   carve((size_t)NN * RR * 4);
  float*  invc  = (float*) carve((size_t)NN * RR * 4);
  __bf16* x1b   = (__bf16*)carve((size_t)NN * HD * 2);
  __bf16* g2    = (__bf16*)carve((size_t)RR * NN * LD * 2);
  float*  oacc  = (float*) carve((size_t)NN * LD * 4);
  float*  out   = (float*)d_out;

  k_cvt_emb<<<(unsigned)(((size_t)NN * KP1 + 255) / 256), 256, 0, stream>>>(emb, embb);
  k_packW1<<<9 * NT1 * KT1, 512, 0, stream>>>(W1, root1, W1p);
  k_packW2<<<9 * NT2 * KT2, 512, 0, stream>>>(W2, root2, W2p);
  k_zero_cnt<<<(NN * RR + 255) / 256, 256, 0, stream>>>(cnt);
  k_count<<<(EE + 255) / 256, 256, 0, stream>>>(dstv, edge_type, cnt);
  k_inv<<<(NN * RR + 255) / 256, 256, 0, stream>>>(cnt, invc);

  dim3 gg((MT + 7) / 8, 1, 9);
  k_gemm1<<<gg, 256, 0, stream>>>(embb, W1p, bias1, h1, x1acc);
  k_scatter1<<<EE / 8, 256, 0, stream>>>(src, dstv, edge_type, h1, invc, x1acc);
  k_relu<<<(NN * HD + 255) / 256, 256, 0, stream>>>(x1acc, x1b);
  k_gemm2<<<gg, 256, 0, stream>>>(x1b, W2p, bias2, g2, oacc);
  k_scatter2<<<EE / 8, 256, 0, stream>>>(src, dstv, edge_type, g2, invc, oacc);
  k_sig<<<(NN * LD + 255) / 256, 256, 0, stream>>>(oacc, out);
}
